// MultiModalFusion_81887846465740
// MI455X (gfx1250) — matrix operand order repaired and checked by reading, not compiled, not run on hardware
//
#include <hip/hip_runtime.h>
#include <hip/hip_bf16.h>

// ---------------------------------------------------------------------------
// CDNA5 (gfx1250, wave32) WMMA helpers
// ---------------------------------------------------------------------------
typedef __bf16 bf16;
typedef __attribute__((ext_vector_type(16))) __bf16 v16bf;
typedef __attribute__((ext_vector_type(8)))  __bf16 v8bf;
typedef __attribute__((ext_vector_type(4)))  __bf16 v4bf;
typedef __attribute__((ext_vector_type(8)))  float  v8f;

#define BS_ROWS 8192   // B*S = 16*512

static __device__ inline bf16 f2bf(float f) {
    return static_cast<bf16>(f);    // native v_cvt (RTNE) on gfx1250
}

static __device__ inline v8f vzero8() {
    v8f z;
#pragma unroll
    for (int i = 0; i < 8; i++) z[i] = 0.0f;
    return z;
}

static __device__ inline v8f wmma_bf(v16bf a, v16bf b, v8f c) {
    return __builtin_amdgcn_wmma_f32_16x16x32_bf16(
        false, a, false, b, (short)0, c, false, false);
}

// A fragment (16x32) from row-major bf16 (ld elements/row, rows 16B aligned):
// lane reads row (l&15), K-runs [k0+ko, +8) and [k0+ko+16, +8), ko = 8*(l>=16)
static __device__ inline v16bf load_a(const bf16* __restrict__ base, int ld,
                                      int k0, int lane) {
    const int m  = lane & 15;
    const int ko = ((lane >> 4) & 1) << 3;
    const bf16* p = base + (size_t)m * ld + k0 + ko;
    v8bf lo = *(const v8bf*)(p);
    v8bf hi = *(const v8bf*)(p + 16);
    return __builtin_shufflevector(lo, hi, 0,1,2,3,4,5,6,7,8,9,10,11,12,13,14,15);
}

// B fragment (32x16) from K-major bf16: element (k,n) at base[n*ld + k]:
// lane reads column n0+(l&15), K-runs [k0+ko, +8) and [k0+ko+16, +8), ko = 8*(l>=16)
static __device__ inline v16bf load_b(const bf16* __restrict__ base, int ld,
                                      int k0, int n0, int lane) {
    const int n  = (lane & 15) + n0;
    const int ko = ((lane >> 4) & 1) << 3;
    const bf16* p = base + (size_t)n * ld + k0 + ko;
    v8bf lo = *(const v8bf*)(p);
    v8bf hi = *(const v8bf*)(p + 16);
    return __builtin_shufflevector(lo, hi, 0,1,2,3,4,5,6,7,8,9,10,11,12,13,14,15);
}

// C/D row for accumulator register r
static __device__ inline int c_row(int lane, int r) {
    return r + (((lane >> 4) & 1) << 3);
}

__device__ __constant__ int c_pi[15] = {0,0,0,0,0,1,1,1,1,2,2,2,3,3,4};
__device__ __constant__ int c_pj[15] = {1,2,3,4,5,2,3,4,5,3,4,5,4,5,5};

// ---------------------------------------------------------------------------
// Weight transpose to bf16 K-major:  Wt[n*K + k] = bf16(W[k*N + n])
// ---------------------------------------------------------------------------
__global__ __launch_bounds__(256)
void k_wtrans(const float* __restrict__ W, bf16* __restrict__ Wt, int K, int N)
{
    int total = K * N;
    for (int i = blockIdx.x * 256 + threadIdx.x; i < total; i += gridDim.x * 256) {
        int k = i / N, n = i - k * N;
        Wt[(size_t)n * K + k] = f2bf(W[i]);
    }
}

// ---------------------------------------------------------------------------
// Stage 1: H = LN(X @ Wm + bm); block = 16 rows x 64 cols, 4 waves
// ---------------------------------------------------------------------------
__global__ __launch_bounds__(128)
void k_proj_ln(const float* __restrict__ X, const bf16* __restrict__ Wt,
               const float* __restrict__ bias, int din,
               const float* __restrict__ g, const float* __restrict__ be,
               float* __restrict__ H32, bf16* __restrict__ Hbf)
{
    __shared__ __align__(16) bf16 xt[16][40];     // one 16x32 bf16 chunk of X
    __shared__ __align__(16) float ht[16][68];
    __shared__ float mu_s[16], rs_s[16];

    const int row0 = blockIdx.x * 16;
    const int tid  = threadIdx.x;
    const int lane = tid & 31;
    const int wv   = tid >> 5;
    const int n0   = wv * 16;

    v8f acc = vzero8();
    for (int k0 = 0; k0 < din; k0 += 32) {
        // stage 16x32 f32 chunk -> bf16 LDS (float4 loads: 8 thr/row x 4 elems)
        {
            int rr = tid >> 3, c4 = (tid & 7) * 4;
            const float4 xv = *(const float4*)(X + (size_t)(row0 + rr) * din + k0 + c4);
            v4bf o;
            o[0] = f2bf(xv.x); o[1] = f2bf(xv.y);
            o[2] = f2bf(xv.z); o[3] = f2bf(xv.w);
            *(v4bf*)&xt[rr][c4] = o;
        }
        __syncthreads();
        v16bf a = load_a(&xt[0][0], 40, 0, lane);
        v16bf b = load_b(Wt, din, k0, n0, lane);
        acc = wmma_bf(a, b, acc);
        __syncthreads();
    }
    {
        int col = (lane & 15) + n0;
        float bv = bias[col];
#pragma unroll
        for (int r = 0; r < 8; r++) ht[c_row(lane, r)][col] = acc[r] + bv;
    }
    __syncthreads();
    if (tid < 16) {
        const float4* rowp = (const float4*)&ht[tid][0];
        float mu = 0.f;
#pragma unroll
        for (int c4 = 0; c4 < 16; c4++) {
            float4 v = rowp[c4];
            mu += (v.x + v.y) + (v.z + v.w);
        }
        mu *= (1.f / 64.f);
        float var = 0.f;
#pragma unroll
        for (int c4 = 0; c4 < 16; c4++) {
            float4 v = rowp[c4];
            float dx = v.x - mu, dy = v.y - mu, dz = v.z - mu, dw = v.w - mu;
            var += (dx * dx + dy * dy) + (dz * dz + dw * dw);
        }
        var *= (1.f / 64.f);
        mu_s[tid] = mu;
        rs_s[tid] = rsqrtf(var + 1e-5f);
    }
    __syncthreads();
    {
        int rr = tid >> 3, c0 = (tid & 7) * 8;
        float mu = mu_s[rr], rs = rs_s[rr];
        float y[8];
        v8bf ob;
#pragma unroll
        for (int u = 0; u < 8; u++) {
            y[u] = (ht[rr][c0 + u] - mu) * rs * g[c0 + u] + be[c0 + u];
            ob[u] = f2bf(y[u]);
        }
        float* hrow = H32 + (size_t)(row0 + rr) * 64 + c0;
        *(float4*)(hrow + 0) = make_float4(y[0], y[1], y[2], y[3]);
        *(float4*)(hrow + 4) = make_float4(y[4], y[5], y[6], y[7]);
        *(v8bf*)(Hbf + (size_t)(row0 + rr) * 64 + c0) = ob;
    }
}

// ---------------------------------------------------------------------------
// Stage 2: Q/K/V.  Q pre-scaled by 0.125.  V stored transposed:
//   Vt[(gr>>9)][col][s]  (per (mod,batch) slab: [64][512])
// ---------------------------------------------------------------------------
__global__ __launch_bounds__(128)
void k_qkv(const bf16* __restrict__ Hbf,
           const bf16* __restrict__ Wqt, const float* __restrict__ bq,
           const bf16* __restrict__ Wkt, const float* __restrict__ bk,
           const bf16* __restrict__ Wvt, const float* __restrict__ bvv,
           bf16* __restrict__ Qbf, bf16* __restrict__ Kbf, bf16* __restrict__ Vt)
{
    const int row0 = blockIdx.x * 16;
    const int tid = threadIdx.x, lane = tid & 31, wv = tid >> 5;
    const int n0 = wv * 16;
    const bf16* Arow = Hbf + (size_t)row0 * 64;
    const bf16* Wts[3] = {Wqt, Wkt, Wvt};
    const float* Bs[3] = {bq, bk, bvv};
    const int col = (lane & 15) + n0;

    v16bf a0 = load_a(Arow, 64, 0,  lane);
    v16bf a1 = load_a(Arow, 64, 32, lane);

#pragma unroll
    for (int t = 0; t < 3; t++) {
        v8f acc = vzero8();
        acc = wmma_bf(a0, load_b(Wts[t], 64, 0,  n0, lane), acc);
        acc = wmma_bf(a1, load_b(Wts[t], 64, 32, n0, lane), acc);
        float bb = Bs[t][col];
        if (t == 0) {
#pragma unroll
            for (int r = 0; r < 8; r++) {
                int row = c_row(lane, r);
                Qbf[(size_t)(row0 + row) * 64 + col] = f2bf((acc[r] + bb) * 0.125f);
            }
        } else if (t == 1) {
#pragma unroll
            for (int r = 0; r < 8; r++) {
                int row = c_row(lane, r);
                Kbf[(size_t)(row0 + row) * 64 + col] = f2bf(acc[r] + bb);
            }
        } else {
#pragma unroll
            for (int r = 0; r < 8; r++) {
                int gr = row0 + c_row(lane, r);
                Vt[((size_t)(gr >> 9) * 64 + col) * 512 + (gr & 511)] = f2bf(acc[r] + bb);
            }
        }
    }
}

// ---------------------------------------------------------------------------
// Stage 3: directed cross-attention.  grid = (30 dirs, 16 batches, 4 qt groups)
// V^T B-fragments are query-tile-invariant -> cached in registers.
// ---------------------------------------------------------------------------
__global__ __launch_bounds__(128)
void k_attn(const bf16* __restrict__ Qbf, const bf16* __restrict__ Kbf,
            const bf16* __restrict__ Vt, float* __restrict__ ctxdir)
{
    __shared__ __align__(16) float scr[16][512];   // 32 KB scores
    __shared__ __align__(16) bf16 pr[16][520];     // 16.6 KB bf16 probs
    __shared__ float red[16][8];
    __shared__ float rmax[16], rsum[16];

    const int d = blockIdx.x;               // 0..29
    const int b = blockIdx.y;               // 0..15
    const int pair = (d < 15) ? d : d - 15;
    const int qi = (d < 15) ? c_pi[pair] : c_pj[pair];
    const int kj = (d < 15) ? c_pj[pair] : c_pi[pair];
    const bf16* Qb  = Qbf + ((size_t)qi * BS_ROWS + (size_t)b * 512) * 64;
    const bf16* Kb  = Kbf + ((size_t)kj * BS_ROWS + (size_t)b * 512) * 64;
    const bf16* Vtb = Vt  + (size_t)(kj * 16 + b) * 64 * 512;
    float* Ob = ctxdir + ((size_t)d * BS_ROWS + (size_t)b * 512) * 64;

    const int tid = threadIdx.x, lane = tid & 31, wv = tid >> 5;
    const int n0 = wv * 16;

    // cache this wave's 16 ctx B-fragments (V^T K-runs) in registers
    v16bf vb[16];
#pragma unroll
    for (int kk = 0; kk < 16; kk++) vb[kk] = load_b(Vtb, 512, kk * 32, n0, lane);

    const int qt0 = blockIdx.z * 8;
    for (int qt = qt0; qt < qt0 + 8; qt++) {
        const int row0 = qt * 16;
        // ---- scores: hoist the two Q A-fragments; each wave: 8 key tiles ----
        {
            v16bf a0 = load_a(Qb + (size_t)row0 * 64, 64, 0,  lane);
            v16bf a1 = load_a(Qb + (size_t)row0 * 64, 64, 32, lane);
            for (int kt = wv; kt < 32; kt += 4) {
                const int key0 = kt * 16;
                v8f acc = vzero8();
                acc = wmma_bf(a0, load_b(Kb + (size_t)key0 * 64, 64, 0,  0, lane), acc);
                acc = wmma_bf(a1, load_b(Kb + (size_t)key0 * 64, 64, 32, 0, lane), acc);
#pragma unroll
                for (int r = 0; r < 8; r++)
                    scr[c_row(lane, r)][key0 + (lane & 15)] = acc[r];
            }
        }
        __syncthreads();
        // ---- softmax (8 threads per row, float4 LDS) -> bf16 probs ----
        {
            const int rr = tid >> 3, seg = tid & 7;
            float4* rowp = (float4*)&scr[rr][seg * 64];
            float lm = -1e30f;
#pragma unroll 4
            for (int c4 = 0; c4 < 16; c4++) {
                float4 v = rowp[c4];
                lm = fmaxf(lm, fmaxf(fmaxf(v.x, v.y), fmaxf(v.z, v.w)));
            }
            red[rr][seg] = lm;
            __syncthreads();
            if (seg == 0) {
                float m2 = red[rr][0];
#pragma unroll
                for (int q = 1; q < 8; q++) m2 = fmaxf(m2, red[rr][q]);
                rmax[rr] = m2;
            }
            __syncthreads();
            float mx = rmax[rr], ls = 0.f;
#pragma unroll 4
            for (int c4 = 0; c4 < 16; c4++) {
                float4 v = rowp[c4];
                v.x = __expf(v.x - mx); v.y = __expf(v.y - mx);
                v.z = __expf(v.z - mx); v.w = __expf(v.w - mx);
                rowp[c4] = v;
                ls += (v.x + v.y) + (v.z + v.w);
            }
            red[rr][seg] = ls;
            __syncthreads();
            if (seg == 0) {
                float s2 = 0.f;
#pragma unroll
                for (int q = 0; q < 8; q++) s2 += red[rr][q];
                rsum[rr] = 1.f / s2;
            }
            __syncthreads();
            float inv = rsum[rr];
            v4bf* prp = (v4bf*)&pr[rr][seg * 64];
#pragma unroll 4
            for (int c4 = 0; c4 < 16; c4++) {
                float4 v = rowp[c4];
                v4bf o;
                o[0] = f2bf(v.x * inv); o[1] = f2bf(v.y * inv);
                o[2] = f2bf(v.z * inv); o[3] = f2bf(v.w * inv);
                prp[c4] = o;
            }
        }
        __syncthreads();
        // ---- ctx = P @ V : A from bf16 LDS, B cached in registers ----
        {
            v8f acc = vzero8();
#pragma unroll
            for (int kk = 0; kk < 16; kk++)
                acc = wmma_bf(load_a(&pr[0][0], 520, kk * 32, lane), vb[kk], acc);
            const int col = (lane & 15) + n0;
#pragma unroll
            for (int r = 0; r < 8; r++) {
                int row = c_row(lane, r);
                Ob[(size_t)(row0 + row) * 64 + col] = acc[r];
            }
        }
        __syncthreads();
    }
}

// ---------------------------------------------------------------------------
// Stage 4: att[p] = (ctx_fwd + ctx_bwd) @ Wo + 2*bo -> E_delta[:, p*64:+64]
// ---------------------------------------------------------------------------
__global__ __launch_bounds__(128)
void k_wo(const float* __restrict__ ctxdir, const bf16* __restrict__ Wot,
          const float* __restrict__ bo, bf16* __restrict__ Ed)
{
    __shared__ __align__(16) bf16 ct[16][72];
    const int row0 = blockIdx.x * 16;
    const int p = blockIdx.y;
    const float* c0 = ctxdir + (size_t)p * BS_ROWS * 64;
    const float* c1 = ctxdir + (size_t)(p + 15) * BS_ROWS * 64;
    const int tid = threadIdx.x, lane = tid & 31, wv = tid >> 5;
    const int n0 = wv * 16;

    // stage (c0 + c1) 16x64 tile as bf16 in LDS
    {
        int rr = tid >> 3, c4 = (tid & 7) * 8;
        size_t ia = (size_t)(row0 + rr) * 64 + c4;
        const float4 x0a = *(const float4*)(c0 + ia);
        const float4 x1a = *(const float4*)(c1 + ia);
        const float4 x0b = *(const float4*)(c0 + ia + 4);
        const float4 x1b = *(const float4*)(c1 + ia + 4);
        v8bf o;
        o[0] = f2bf(x0a.x + x1a.x); o[1] = f2bf(x0a.y + x1a.y);
        o[2] = f2bf(x0a.z + x1a.z); o[3] = f2bf(x0a.w + x1a.w);
        o[4] = f2bf(x0b.x + x1b.x); o[5] = f2bf(x0b.y + x1b.y);
        o[6] = f2bf(x0b.z + x1b.z); o[7] = f2bf(x0b.w + x1b.w);
        *(v8bf*)&ct[rr][c4] = o;
    }
    __syncthreads();

    v8f acc = vzero8();
    acc = wmma_bf(load_a(&ct[0][0], 72, 0,  lane), load_b(Wot, 64, 0,  n0, lane), acc);
    acc = wmma_bf(load_a(&ct[0][0], 72, 32, lane), load_b(Wot, 64, 32, n0, lane), acc);
    const int col = (lane & 15) + n0;
    const float bb = 2.0f * bo[col];
#pragma unroll
    for (int r = 0; r < 8; r++) {
        int row = c_row(lane, r);
        Ed[(size_t)(row0 + row) * 960 + p * 64 + col] = f2bf(acc[r] + bb);
    }
}

// ---------------------------------------------------------------------------
// Stage 5: h = relu(LN(E_delta @ W1 + b1)); each wave: 6 N-tiles (384 cols)
// ---------------------------------------------------------------------------
__global__ __launch_bounds__(128)
void k_mlp1(const bf16* __restrict__ Ed, const bf16* __restrict__ W1t,
            const float* __restrict__ b1, const float* __restrict__ g,
            const float* __restrict__ be, bf16* __restrict__ hbf)
{
    __shared__ __align__(16) float ht[16][388];
    __shared__ float mu_s[16], rs_s[16];

    const int row0 = blockIdx.x * 16;
    const int tid = threadIdx.x, lane = tid & 31, wv = tid >> 5;
    const bf16* Arow = Ed + (size_t)row0 * 960;

    v8f acc[6];
#pragma unroll
    for (int t = 0; t < 6; t++) acc[t] = vzero8();

    for (int k0 = 0; k0 < 960; k0 += 32) {
        v16bf a = load_a(Arow, 960, k0, lane);
#pragma unroll
        for (int t = 0; t < 6; t++) {
            v16bf b = load_b(W1t, 960, k0, t * 64 + wv * 16, lane);
            acc[t] = wmma_bf(a, b, acc[t]);
        }
    }
#pragma unroll
    for (int t = 0; t < 6; t++) {
        int col = (lane & 15) + t * 64 + wv * 16;
        float bb = b1[col];
#pragma unroll
        for (int r = 0; r < 8; r++) ht[c_row(lane, r)][col] = acc[t][r] + bb;
    }
    __syncthreads();
    if (tid < 16) {
        const float4* rowp = (const float4*)&ht[tid][0];
        float mu = 0.f;
        for (int c4 = 0; c4 < 96; c4++) {
            float4 v = rowp[c4];
            mu += (v.x + v.y) + (v.z + v.w);
        }
        mu *= (1.f / 384.f);
        float var = 0.f;
        for (int c4 = 0; c4 < 96; c4++) {
            float4 v = rowp[c4];
            float dx = v.x - mu, dy = v.y - mu, dz = v.z - mu, dw = v.w - mu;
            var += (dx * dx + dy * dy) + (dz * dz + dw * dw);
        }
        var *= (1.f / 384.f);
        mu_s[tid] = mu;
        rs_s[tid] = rsqrtf(var + 1e-5f);
    }
    __syncthreads();
    // 128 threads, 16 rows x 384 cols: 8 threads/row, 48 cols each
    {
        int rr = tid >> 3, c0 = (tid & 7) * 48;
        float mu = mu_s[rr], rs = rs_s[rr];
#pragma unroll
        for (int u = 0; u < 12; u++) {
            int cc = c0 + u * 4;
            float4 v = *(const float4*)&ht[rr][cc];
            v4bf o;
            o[0] = f2bf(fmaxf((v.x - mu) * rs * g[cc + 0] + be[cc + 0], 0.f));
            o[1] = f2bf(fmaxf((v.y - mu) * rs * g[cc + 1] + be[cc + 1], 0.f));
            o[2] = f2bf(fmaxf((v.z - mu) * rs * g[cc + 2] + be[cc + 2], 0.f));
            o[3] = f2bf(fmaxf((v.w - mu) * rs * g[cc + 3] + be[cc + 3], 0.f));
            *(v4bf*)(hbf + (size_t)(row0 + rr) * 384 + cc) = o;
        }
    }
}

// ---------------------------------------------------------------------------
// Stage 6: out = LN(h @ W2 + b2 + sum_m H[m], fin)
// ---------------------------------------------------------------------------
__global__ __launch_bounds__(128)
void k_mlp2(const bf16* __restrict__ hbf, const bf16* __restrict__ W2t,
            const float* __restrict__ b2, const float* __restrict__ H32,
            const float* __restrict__ g, const float* __restrict__ be,
            float* __restrict__ out)
{
    __shared__ __align__(16) float ht[16][68];
    __shared__ float mu_s[16], rs_s[16];

    const int row0 = blockIdx.x * 16;
    const int tid = threadIdx.x, lane = tid & 31, wv = tid >> 5;
    const int n0 = wv * 16;
    const bf16* Arow = hbf + (size_t)row0 * 384;

    v8f acc = vzero8();
#pragma unroll
    for (int k0 = 0; k0 < 384; k0 += 32) {
        v16bf a = load_a(Arow, 384, k0, lane);
        v16bf b = load_b(W2t, 384, k0, n0, lane);
        acc = wmma_bf(a, b, acc);
    }
    const int col = (lane & 15) + n0;
    const float bb = b2[col];
#pragma unroll
    for (int r = 0; r < 8; r++) {
        int row = c_row(lane, r);
        float resid = 0.f;
#pragma unroll
        for (int mmod = 0; mmod < 6; mmod++)
            resid += H32[((size_t)mmod * BS_ROWS + row0 + row) * 64 + col];
        ht[row][col] = acc[r] + bb + resid;
    }
    __syncthreads();
    if (tid < 16) {
        const float4* rowp = (const float4*)&ht[tid][0];
        float mu = 0.f;
#pragma unroll
        for (int c4 = 0; c4 < 16; c4++) {
            float4 v = rowp[c4];
            mu += (v.x + v.y) + (v.z + v.w);
        }
        mu *= (1.f / 64.f);
        float var = 0.f;
#pragma unroll
        for (int c4 = 0; c4 < 16; c4++) {
            float4 v = rowp[c4];
            float dx = v.x - mu, dy = v.y - mu, dz = v.z - mu, dw = v.w - mu;
            var += (dx * dx + dy * dy) + (dz * dz + dw * dw);
        }
        var *= (1.f / 64.f);
        mu_s[tid] = mu;
        rs_s[tid] = rsqrtf(var + 1e-5f);
    }
    __syncthreads();
    {
        int rr = tid >> 3, c0 = (tid & 7) * 8;
        float mu = mu_s[rr], rs = rs_s[rr];
        float y[8];
#pragma unroll
        for (int u = 0; u < 8; u++)
            y[u] = (ht[rr][c0 + u] - mu) * rs * g[c0 + u] + be[c0 + u];
        float* orow = out + (size_t)(row0 + rr) * 64 + c0;
        *(float4*)(orow + 0) = make_float4(y[0], y[1], y[2], y[3]);
        *(float4*)(orow + 4) = make_float4(y[4], y[5], y[6], y[7]);
    }
}

// ---------------------------------------------------------------------------
// Host orchestration
// ---------------------------------------------------------------------------
extern "C" void kernel_launch(void* const* d_in, const int* in_sizes, int n_in,
                              void* d_out, int out_size, void* d_ws, size_t ws_size,
                              hipStream_t stream)
{
    (void)in_sizes; (void)n_in; (void)out_size; (void)ws_size;

    const float* X[6];
    const float* Wm[6];
    const float* bm[6];
    for (int m = 0; m < 6; m++) {
        X[m]  = (const float*)d_in[m];
        Wm[m] = (const float*)d_in[6 + 2 * m];
        bm[m] = (const float*)d_in[7 + 2 * m];
    }
    static const int dins[6] = {768, 512, 256, 256, 256, 768};
    const float* ln_g = (const float*)d_in[18];
    const float* ln_b = (const float*)d_in[19];
    const float* Wq = (const float*)d_in[20];  const float* bq = (const float*)d_in[21];
    const float* Wk = (const float*)d_in[22];  const float* bk = (const float*)d_in[23];
    const float* Wv = (const float*)d_in[24];  const float* bvv = (const float*)d_in[25];
    const float* Wo = (const float*)d_in[26];  const float* bo = (const float*)d_in[27];
    const float* W1 = (const float*)d_in[28];  const float* b1 = (const float*)d_in[29];
    const float* ln1_g = (const float*)d_in[30]; const float* ln1_b = (const float*)d_in[31];
    const float* W2 = (const float*)d_in[32];  const float* b2 = (const float*)d_in[33];
    const float* fin_g = (const float*)d_in[34]; const float* fin_b = (const float*)d_in[35];

    char* ws = (char*)d_ws;
    size_t off = 0;
    auto salloc = [&](size_t bytes) -> void* {
        void* p = ws + off;
        off += (bytes + 255) & ~(size_t)255;
        return p;
    };
    float* H32    = (float*)salloc((size_t)6  * BS_ROWS * 64 * sizeof(float));
    bf16*  Hbf    = (bf16*) salloc((size_t)6  * BS_ROWS * 64 * sizeof(bf16));
    bf16*  Qbf    = (bf16*) salloc((size_t)6  * BS_ROWS * 64 * sizeof(bf16));
    bf16*  Kbf    = (bf16*) salloc((size_t)6  * BS_ROWS * 64 * sizeof(bf16));
    bf16*  Vt     = (bf16*) salloc((size_t)6  * BS_ROWS * 64 * sizeof(bf16));
    float* ctxdir = (float*)salloc((size_t)30 * BS_ROWS * 64 * sizeof(float));
    bf16*  Ed     = (bf16*) salloc((size_t)BS_ROWS * 960 * sizeof(bf16));
    bf16*  hb     = (bf16*) salloc((size_t)BS_ROWS * 384 * sizeof(bf16));
    bf16* Wmt[6];
    for (int m = 0; m < 6; m++)
        Wmt[m] = (bf16*)salloc((size_t)64 * dins[m] * sizeof(bf16));
    bf16* Wqt = (bf16*)salloc((size_t)64 * 64 * sizeof(bf16));
    bf16* Wkt = (bf16*)salloc((size_t)64 * 64 * sizeof(bf16));
    bf16* Wvt = (bf16*)salloc((size_t)64 * 64 * sizeof(bf16));
    bf16* Wot = (bf16*)salloc((size_t)64 * 64 * sizeof(bf16));
    bf16* W1t = (bf16*)salloc((size_t)960 * 384 * sizeof(bf16));
    bf16* W2t = (bf16*)salloc((size_t)384 * 64 * sizeof(bf16));

    auto trans = [&](const float* W, bf16* Wt, int K, int N) {
        int blocks = (K * N + 255) / 256;
        k_wtrans<<<blocks, 256, 0, stream>>>(W, Wt, K, N);
    };
    for (int m = 0; m < 6; m++) trans(Wm[m], Wmt[m], dins[m], 64);
    trans(Wq, Wqt, 64, 64);
    trans(Wk, Wkt, 64, 64);
    trans(Wv, Wvt, 64, 64);
    trans(Wo, Wot, 64, 64);
    trans(W1, W1t, 960, 384);
    trans(W2, W2t, 384, 64);

    // Stage 1: projections + LN
    for (int m = 0; m < 6; m++) {
        k_proj_ln<<<BS_ROWS / 16, 128, 0, stream>>>(
            X[m], Wmt[m], bm[m], dins[m], ln_g, ln_b,
            H32 + (size_t)m * BS_ROWS * 64, Hbf + (size_t)m * BS_ROWS * 64);
    }
    // Stage 2: Q/K/V
    k_qkv<<<6 * BS_ROWS / 16, 128, 0, stream>>>(
        Hbf, Wqt, bq, Wkt, bk, Wvt, bvv, Qbf, Kbf, Vt);
    // Stage 3: 30 directed attentions x 16 batches x 4 query-tile groups
    k_attn<<<dim3(30, 16, 4), 128, 0, stream>>>(Qbf, Kbf, Vt, ctxdir);
    // Stage 4: Wo + concat into E_delta
    k_wo<<<dim3(BS_ROWS / 16, 15), 128, 0, stream>>>(ctxdir, Wot, bo, Ed);
    // Stage 5: W1 + LN + ReLU
    k_mlp1<<<BS_ROWS / 16, 128, 0, stream>>>(Ed, W1t, b1, ln1_g, ln1_b, hb);
    // Stage 6: W2 + residual + final LN
    k_mlp2<<<BS_ROWS / 16, 128, 0, stream>>>(hb, W2t, b2, H32, fin_g, fin_b,
                                             (float*)d_out);
}
